// MultiHeadAttention_dis_bias_62088047231275
// MI455X (gfx1250) — matrix operand order repaired and checked by reading, not compiled, not run on hardware
//
#include <hip/hip_runtime.h>

// MI455X / gfx1250, wave32. All matrix math via v_wmma_f32_16x16x32_bf16.
typedef __attribute__((ext_vector_type(16))) __bf16 v16bf;
typedef __attribute__((ext_vector_type(8)))  float  v8f;

#define WMMA_BF16(a, b, c) \
  __builtin_amdgcn_wmma_f32_16x16x32_bf16(false, (a), false, (b), (short)0, (c), false, false)

// Problem constants
// B=4, N=512, C=128, ZC=8, BUCKETS=20, DIS_END=15

// ---------------------------------------------------------------------------
// kP: prep — Pt[b,c,i] = bf16(pocket_rep[b,i,c]); WoT[co,k] = bf16(Wo[k,co]);
//     zero the 8 affinity/prmsd accumulators.
// ---------------------------------------------------------------------------
__global__ __launch_bounds__(256) void kP(const float* __restrict__ pocket,
                                          const float* __restrict__ Wo,
                                          __bf16* __restrict__ Pt,
                                          __bf16* __restrict__ WoT,
                                          float* __restrict__ sums) {
  int idx = blockIdx.x * 256 + threadIdx.x;
  if (idx < 262144) {                       // Pt: 4*128*512
    int b = idx >> 16, rem = idx & 65535;
    int c = rem >> 9, i = rem & 511;
    Pt[idx] = (__bf16)pocket[((size_t)b * 512 + i) * 128 + c];
  } else if (idx < 262144 + 131072) {       // WoT: 128*1024
    int t = idx - 262144;
    int co = t >> 10, k = t & 1023;
    WoT[t] = (__bf16)Wo[k * 128 + co];
  } else if (idx < 262144 + 131072 + 8) {
    sums[idx - 393216] = 0.f;
  }
}

// ---------------------------------------------------------------------------
// kA: one block per (b, j). The single pass over z (537 MB => the roofline).
//   phase0: WlT (bf16, zero-padded to 16 cols) + distance-bias tile into LDS
//   phase1: proj[i,zc] = z[b,i,j,:] @ Wl  via WMMA bf16 (A from global, B from LDS)
//   phase2: softmax over i per zc, + dis-bias (view-remap), mask z_mask[zc&3],
//           write zt bf16 [b,zc,j,i]
//   phase3: affinity / prmsd partial sums -> atomicAdd
// ---------------------------------------------------------------------------
__global__ __launch_bounds__(256) void kA(
    const float* __restrict__ z, const unsigned char* __restrict__ zmask,
    const float* __restrict__ dis,
    const float* __restrict__ Wl, const float* __restrict__ bl,
    const float* __restrict__ Wd, const float* __restrict__ bd,
    const float* __restrict__ Wg, const float* __restrict__ bg,
    const float* __restrict__ We, const float* __restrict__ be,
    const float* __restrict__ Wgp, const float* __restrict__ bgp,
    const float* __restrict__ Wep, const float* __restrict__ bep,
    __bf16* __restrict__ zt, float* __restrict__ sums) {
  __shared__ float  s_proj[512 * 8];   // 16 KB: proj tile / final zt tile (f32)
  __shared__ float  s_demb[8 * 512];   // 16 KB: dis-bias db[zc][i] for this (b,j)
  __shared__ __bf16 s_wlt[16 * 128];   // 4 KB : Wl^T, cols 8..15 zero
  __shared__ float  s_wred[16];

  const int tid  = threadIdx.x;
  const int lane = tid & 31;
  const int wv   = tid >> 5;
  const int j = blockIdx.x, b = blockIdx.y;

  // ---- phase 0a: Wl^T into LDS (B-operand: per-lane contiguous K)
  for (int idx = tid; idx < 16 * 128; idx += 256) {
    int col = idx >> 7, k = idx & 127;
    s_wlt[idx] = (__bf16)((col < 8) ? Wl[k * 8 + col] : 0.f);
  }
  // ---- phase 0b: distance bucket bias, faithful raw-view remap:
  //   db[b,zc,j,i] = Demb[b, zc*64+(j>>3), (j&7)*64+(i>>3), i&7]
  {
    const float stepv = 15.f / 21.f;
    const float KC = 1.14136f * 7.38905609893065f;  // 1.14136 * e^2
    for (int p = tid; p < 512; p += 256) {
      int zc = p >> 6, n2l = p & 63;
      int n1 = zc * 64 + (j >> 3);
      int n2 = (j & 7) * 64 + n2l;
      float d = dis[((size_t)b * 512 + n1) * 512 + n2];
      float x = d / stepv;
      float f[20];
#pragma unroll
      for (int v = 0; v < 20; ++v) {
        float diff = x - (float)(v + 1);
        float t0 = diff + 1.f, t1 = 1.f - diff;
        float s0 = (t0 > 0.f) ? __expf(-1.f / t0) : 0.f;
        float s1 = (t1 > 0.f) ? __expf(-1.f / t1) : 0.f;
        f[v] = KC * s0 * s1;
      }
#pragma unroll
      for (int k = 0; k < 8; ++k) {
        float acc = bd[k];
#pragma unroll
        for (int v = 0; v < 20; ++v) acc += f[v] * Wd[v * 8 + k];
        s_demb[zc * 512 + n2l * 8 + k] = acc;
      }
    }
  }
  __syncthreads();

  // ---- phase 1: proj = z @ Wl (M=i tile of 16, N=zc(16, 8 live), K=c 128)
  {
    const int hi = lane >> 4, lo = lane & 15;
    for (int t16 = 0; t16 < 4; ++t16) {
      const int i0 = wv * 64 + t16 * 16;
      const float* rp = z + (((size_t)b * 512 + (i0 + lo)) * 512 + j) * 128;
      v8f acc = {};
#pragma unroll
      for (int kt = 0; kt < 4; ++kt) {
        int ka = kt * 32 + hi * 16;
        v16bf A;
#pragma unroll
        for (int e = 0; e < 8; ++e) A[e] = (__bf16)rp[ka + e];
#pragma unroll
        for (int e = 0; e < 8; ++e) A[8 + e] = (__bf16)rp[ka + 8 + e];
        v16bf Bm;
        __builtin_memcpy(&Bm, &s_wlt[lo * 128 + kt * 32 + hi * 16], 32);
        acc = WMMA_BF16(A, Bm, acc);
      }
      if (lo < 8) {                         // D: vgpr d -> M=d+8*hi, N=lo
        float blv = bl[lo];
#pragma unroll
        for (int d = 0; d < 8; ++d)
          s_proj[(i0 + d + hi * 8) * 8 + lo] = acc[d] + blv;
      }
    }
  }
  __syncthreads();

  // ---- phase 2: softmax over i (per zc = warp), + bias, mask, write zt bf16
  {
    const int zc = wv;
    float m = -3.0e38f;
    for (int t = 0; t < 16; ++t) m = fmaxf(m, s_proj[(lane + 32 * t) * 8 + zc]);
#pragma unroll
    for (int o = 16; o > 0; o >>= 1) m = fmaxf(m, __shfl_xor(m, o, 32));
    float s = 0.f;
    for (int t = 0; t < 16; ++t) s += __expf(s_proj[(lane + 32 * t) * 8 + zc] - m);
#pragma unroll
    for (int o = 16; o > 0; o >>= 1) s += __shfl_xor(s, o, 32);
    float inv = 1.f / s;
    // mask_rep bug: mask comes from z_mask[(b*8+zc)%4] = z_mask[zc&3]
    const unsigned char* mrow = zmask + ((size_t)(zc & 3) * 512 + j) * 512;
    __bf16* orow = zt + (((size_t)b * 8 + zc) * 512 + j) * 512;
    for (int t = 0; t < 16; ++t) {
      int i = lane + 32 * t;
      float v = __expf(s_proj[i * 8 + zc] - m) * inv + s_demb[zc * 512 + i];
      if (mrow[i]) v = 1e-9f;
      s_proj[i * 8 + zc] = v;
      orow[i] = (__bf16)v;
    }
  }
  __syncthreads();

  // ---- phase 3: affinity / prmsd partial sums (true per-b mask here)
  {
    float wg[8], wE[8], wgp[8], wep[8];
#pragma unroll
    for (int k = 0; k < 8; ++k) {
      wg[k] = Wg[k]; wE[k] = We[k]; wgp[k] = Wgp[k]; wep[k] = Wep[k];
    }
    float bgv = bg[0], bev = be[0], bgpv = bgp[0], bepv = bep[0];
    float pa = 0.f, pp = 0.f;
    for (int i = tid; i < 512; i += 256) {
      if (zmask[((size_t)b * 512 + i) * 512 + j]) {
        float dg = bgv, de = bev, dgp = bgpv, dep = bepv;
#pragma unroll
        for (int k = 0; k < 8; ++k) {
          float v = s_proj[i * 8 + k];
          dg += v * wg[k]; de += v * wE[k]; dgp += v * wgp[k]; dep += v * wep[k];
        }
        pa += de / (1.f + __expf(-dg));
        pp += dep / (1.f + __expf(-dgp));
      }
    }
#pragma unroll
    for (int o = 16; o > 0; o >>= 1) {
      pa += __shfl_xor(pa, o, 32);
      pp += __shfl_xor(pp, o, 32);
    }
    if (lane == 0) { s_wred[wv] = pa; s_wred[8 + wv] = pp; }
    __syncthreads();
    if (tid == 0) {
      float ta = 0.f, tp = 0.f;
      for (int w = 0; w < 8; ++w) { ta += s_wred[w]; tp += s_wred[8 + w]; }
      atomicAdd(&sums[b], ta);
      atomicAdd(&sums[4 + b], tp);
    }
  }
}

// ---------------------------------------------------------------------------
// kB: lig[b, zc*N+j, c] = sum_i zt[b,zc,j,i] * pocket_rep[b,i,c]
//     written already transposed as lig[b, j, zc*128 + c] (bf16). K=512.
// ---------------------------------------------------------------------------
__global__ __launch_bounds__(256) void kB(const __bf16* __restrict__ zt,
                                          const __bf16* __restrict__ Pt,
                                          __bf16* __restrict__ lig) {
  const int tid = threadIdx.x, lane = tid & 31, wv = tid >> 5;
  const int jt = blockIdx.x, zc = blockIdx.y, b = blockIdx.z;
  const int j0 = jt * 16;
  const int hi = lane >> 4, lo = lane & 15;
  const __bf16* arow = zt + (((size_t)b * 8 + zc) * 512 + (j0 + lo)) * 512;
  const __bf16* brow = Pt + ((size_t)b * 128 + wv * 16 + lo) * 512;
  v8f acc = {};
  for (int kt = 0; kt < 16; ++kt) {
    v16bf A, Bm;
    int ka = kt * 32 + hi * 16;
    __builtin_memcpy(&A, arow + ka, 32);
    __builtin_memcpy(&Bm, brow + kt * 32 + hi * 16, 32);
    acc = WMMA_BF16(A, Bm, acc);
  }
#pragma unroll
  for (int d = 0; d < 8; ++d) {
    int jr = j0 + d + hi * 8;
    int col = zc * 128 + wv * 16 + lo;
    lig[((size_t)b * 512 + jr) * 1024 + col] = (__bf16)acc[d];
  }
}

// ---------------------------------------------------------------------------
// kC: out[b,j,:] = lig[b,j,:] @ Wo + bo   (K=1024, N=128), f32 output.
// ---------------------------------------------------------------------------
__global__ __launch_bounds__(256) void kC(const __bf16* __restrict__ lig,
                                          const __bf16* __restrict__ WoT,
                                          const float* __restrict__ bo,
                                          float* __restrict__ out) {
  const int tid = threadIdx.x, lane = tid & 31, wv = tid >> 5;
  const int jt = blockIdx.x, b = blockIdx.y;
  const int j0 = jt * 16;
  const int hi = lane >> 4, lo = lane & 15;
  const __bf16* arow = lig + ((size_t)b * 512 + j0 + lo) * 1024;
  const __bf16* brow = WoT + (size_t)(wv * 16 + lo) * 1024;
  v8f acc = {};
  for (int kt = 0; kt < 32; ++kt) {
    v16bf A, Bm;
    int ka = kt * 32 + hi * 16;
    __builtin_memcpy(&A, arow + ka, 32);
    __builtin_memcpy(&Bm, brow + kt * 32 + hi * 16, 32);
    acc = WMMA_BF16(A, Bm, acc);
  }
  int co = wv * 16 + lo;
  float bov = bo[co];
#pragma unroll
  for (int d = 0; d < 8; ++d) {
    int jr = j0 + d + hi * 8;
    out[((size_t)b * 512 + jr) * 128 + co] = acc[d] + bov;
  }
}

// ---------------------------------------------------------------------------
// kF: leaky-ReLU finalize of affinity / prmsd into the output tail.
// ---------------------------------------------------------------------------
__global__ void kF(const float* __restrict__ sums, const float* __restrict__ bias,
                   const float* __restrict__ bias_p, float* __restrict__ out) {
  int t = threadIdx.x;
  if (t < 4) {
    float x = bias[0] + sums[t];
    out[262144 + t] = (x > 0.f) ? x : 0.01f * x;
  } else if (t < 8) {
    int bb = t - 4;
    float x = bias_p[0] + sums[4 + bb];
    out[262148 + bb] = (x > 0.f) ? x : 0.01f * x;
  }
}

extern "C" void kernel_launch(void* const* d_in, const int* in_sizes, int n_in,
                              void* d_out, int out_size, void* d_ws, size_t ws_size,
                              hipStream_t stream) {
  (void)in_sizes; (void)n_in; (void)out_size; (void)ws_size;
  const float* z    = (const float*)d_in[0];
  const unsigned char* zm = (const unsigned char*)d_in[1];  // bool mask
  const float* pr   = (const float*)d_in[2];
  const float* dis  = (const float*)d_in[3];
  const float* Wl   = (const float*)d_in[4];
  const float* bl   = (const float*)d_in[5];
  const float* Wd   = (const float*)d_in[6];
  const float* bd   = (const float*)d_in[7];
  const float* Wg   = (const float*)d_in[8];
  const float* bg   = (const float*)d_in[9];
  const float* We   = (const float*)d_in[10];
  const float* be   = (const float*)d_in[11];
  const float* Wgp  = (const float*)d_in[12];
  const float* bgp  = (const float*)d_in[13];
  const float* Wep  = (const float*)d_in[14];
  const float* bep  = (const float*)d_in[15];
  const float* bias = (const float*)d_in[16];
  const float* biasp= (const float*)d_in[17];
  const float* Wo   = (const float*)d_in[18];
  const float* bo   = (const float*)d_in[19];
  float* out = (float*)d_out;

  char* ws = (char*)d_ws;                       // workspace layout (bytes):
  __bf16* zt  = (__bf16*)(ws);                  // [4,8,512,512] bf16 : 16 MB
  __bf16* Pt  = (__bf16*)(ws + 16777216);       // [4,128,512]   bf16 : 512 KB
  __bf16* WoT = (__bf16*)(ws + 17301504);       // [128,1024]    bf16 : 256 KB
  __bf16* lig = (__bf16*)(ws + 17563648);       // [4,512,1024]  bf16 : 4 MB
  float*  sums= (float*)(ws + 21757952);        // 8 f32 accumulators

  kP<<<1537, 256, 0, stream>>>(pr, Wo, Pt, WoT, sums);
  kA<<<dim3(512, 4), 256, 0, stream>>>(z, zm, dis, Wl, bl, Wd, bd, Wg, bg,
                                       We, be, Wgp, bgp, Wep, bep, zt, sums);
  kB<<<dim3(32, 8, 4), 256, 0, stream>>>(zt, Pt, lig);
  kC<<<dim3(32, 4), 256, 0, stream>>>(lig, WoT, bo, out);
  kF<<<1, 32, 0, stream>>>(sums, bias, biasp, out);
}
